// TransformerEncoder_45148696216132
// MI455X (gfx1250) — matrix operand order repaired and checked by reading, not compiled, not run on hardware
//
#include <hip/hip_runtime.h>
#include <hip/hip_bf16.h>

// ---------------------------------------------------------------------------
// Transformer encoder (B=2,S=2048,D=768,H=12,hd=64,I=3072,L=4), fp32 ref.
// bf16 WMMA (v_wmma_f32_16x16x32_bf16) for all matmuls, fp32 accumulate.
// Double-buffered LDS staging; async global->LDS copies (ASYNCcnt) for the
// straight-copy tiles. Residual stream lives in d_out (fp32).
// ---------------------------------------------------------------------------

typedef __attribute__((ext_vector_type(16))) __bf16 v16bf;
typedef __attribute__((ext_vector_type(8)))  float  v8f;
typedef __attribute__((ext_vector_type(4)))  int    i4;

#define D_   768
#define H_   12
#define HD_  64
#define S_   2048
#define B_   2
#define L_   4
#define I_   3072
#define MTOK 4096   // B_*S_

#if defined(__gfx1250__) && __has_builtin(__builtin_amdgcn_global_load_async_to_lds_b128)
#define HAVE_ASYNC 1
#else
#define HAVE_ASYNC 0
#endif

#if HAVE_ASYNC
typedef __attribute__((address_space(1))) i4 gi4;
typedef __attribute__((address_space(3))) i4 li4;
#endif

__device__ __forceinline__ void async_b128(const void* g, void* l) {
#if HAVE_ASYNC
  __builtin_amdgcn_global_load_async_to_lds_b128((gi4*)g, (li4*)l, 0, 0);
#else
  *(uint4*)l = *(const uint4*)g;
#endif
}

__device__ __forceinline__ void wait_async() {
#if HAVE_ASYNC
#if __has_builtin(__builtin_amdgcn_s_wait_asynccnt)
  __builtin_amdgcn_s_wait_asynccnt(0);
#else
  asm volatile("s_wait_asynccnt 0x0" ::: "memory");
#endif
#endif
}

__device__ __forceinline__ unsigned short f2bf(float f) {
  unsigned int u = __float_as_uint(f);
  unsigned int r = u + 0x7FFFu + ((u >> 16) & 1u);   // round-to-nearest-even
  return (unsigned short)(r >> 16);
}

__device__ __forceinline__ unsigned int pack_bf2(float a, float b) {
  return (unsigned int)f2bf(a) | ((unsigned int)f2bf(b) << 16);
}

// Load a 16-half WMMA fragment as two 16B chunks (ds_load_b128 / global b128).
__device__ __forceinline__ v16bf ld_frag(const unsigned short* p0,
                                         const unsigned short* p1) {
  union { v16bf v; uint4 q[2]; } u;
  u.q[0] = *(const uint4*)p0;
  u.q[1] = *(const uint4*)p1;
  return u.v;
}

__device__ __forceinline__ v8f wmma_bf16(v16bf a, v16bf b, v8f c) {
  return __builtin_amdgcn_wmma_f32_16x16x32_bf16(false, a, false, b,
                                                 (short)0, c, false, false);
}

// ---------------------------------------------------------------------------
// LayerNorm: fp32 [row, 768] -> bf16 [row, 768], per-row mean/var.
// ---------------------------------------------------------------------------
__global__ __launch_bounds__(256) void ln_kernel(const float* __restrict__ x,
                                                 const float* __restrict__ w,
                                                 const float* __restrict__ b,
                                                 unsigned short* __restrict__ out) {
  __shared__ float rs[8], rss[8];
  const int row = blockIdx.x;
  const int t = threadIdx.x;
  const float* xr = x + (size_t)row * D_;
  float v0 = xr[t], v1 = xr[t + 256], v2 = xr[t + 512];
  float s = v0 + v1 + v2;
  float ss = v0 * v0 + v1 * v1 + v2 * v2;
#pragma unroll
  for (int d = 16; d >= 1; d >>= 1) {
    s += __shfl_xor(s, d);
    ss += __shfl_xor(ss, d);
  }
  if ((t & 31) == 0) { rs[t >> 5] = s; rss[t >> 5] = ss; }
  __syncthreads();
  float tot = 0.f, tots = 0.f;
#pragma unroll
  for (int i = 0; i < 8; i++) { tot += rs[i]; tots += rss[i]; }
  const float mu = tot * (1.0f / 768.0f);
  const float var = tots * (1.0f / 768.0f) - mu * mu;
  const float inv = rsqrtf(var + 1e-5f);
  unsigned short* orow = out + (size_t)row * D_;
  orow[t]       = f2bf((v0 - mu) * inv * w[t]       + b[t]);
  orow[t + 256] = f2bf((v1 - mu) * inv * w[t + 256] + b[t + 256]);
  orow[t + 512] = f2bf((v2 - mu) * inv * w[t + 512] + b[t + 512]);
}

// ---------------------------------------------------------------------------
// WMMA GEMM: C[M,N] = A_bf16[M,K] @ B_fp32[K,N] + bias, block tile 128x128,
// 8 waves of 32(M)x64(N), K-chunks of 32, double-buffered LDS.
// A staged via async global->LDS copies; B converted fp32->bf16 on the fly,
// stored column-major so B fragments are two contiguous b128 LDS loads.
// EPI 0: store bf16.  EPI 1: exact GELU then bf16.  EPI 2: fp32 out += val.
// ---------------------------------------------------------------------------
template <int EPI>
__global__ __launch_bounds__(256) void gemm_kernel(
    const unsigned short* __restrict__ A, const float* __restrict__ Bw,
    const float* __restrict__ bias, void* __restrict__ outp, int N, int K) {
  __shared__ unsigned short Al[2][128 * 40];  // row-major, pitch 40 halfs
  __shared__ unsigned short Bl[2][128 * 40];  // column-major, pitch 40 halfs

  const int tid = threadIdx.x;
  const int lane = tid & 31, wid = tid >> 5;
  const int r = lane & 15, hl = lane >> 4;
  const int m_base = blockIdx.y * 128;
  const int n_base = blockIdx.x * 128;
  const int m_w = (wid & 3) * 32;
  const int n_w = (wid >> 2) * 64;

  v8f acc[2][4];
#pragma unroll
  for (int a = 0; a < 2; a++)
#pragma unroll
    for (int bi = 0; bi < 4; bi++)
#pragma unroll
      for (int v = 0; v < 8; v++) acc[a][bi][v] = 0.0f;

  const int arow = tid >> 1, aseg = tid & 1;  // A copy: 128 rows x 2 segs
  const int bc0 = (tid & 31) * 4;             // B copy: 32 col-groups of 4
  const int bkp0 = (tid >> 5) * 2;            // 8 groups x 2 k-pairs
  const int NC = K / 32;

  const unsigned short* Ag = A + (size_t)(m_base + arow) * K + aseg * 16;

  auto stageA = [&](int kc, int buf) {
    const unsigned short* g = Ag + kc * 32;
    unsigned short* l = &Al[buf][arow * 40 + aseg * 16];
    async_b128(g, l);
    async_b128(g + 8, l + 8);
  };
  auto loadB = [&](int kc, float4* f) {
#pragma unroll
    for (int it = 0; it < 2; it++) {
      const int kp = bkp0 + it;
      const float* g0 = Bw + (size_t)(kc * 32 + 2 * kp) * N + n_base + bc0;
      f[2 * it]     = *(const float4*)g0;
      f[2 * it + 1] = *(const float4*)(g0 + N);
    }
  };
  auto storeB = [&](int buf, const float4* f) {
#pragma unroll
    for (int it = 0; it < 2; it++) {
      const int kp = bkp0 + it;
      const float4 f0 = f[2 * it], f1 = f[2 * it + 1];
      *(unsigned int*)&Bl[buf][(bc0 + 0) * 40 + 2 * kp] = pack_bf2(f0.x, f1.x);
      *(unsigned int*)&Bl[buf][(bc0 + 1) * 40 + 2 * kp] = pack_bf2(f0.y, f1.y);
      *(unsigned int*)&Bl[buf][(bc0 + 2) * 40 + 2 * kp] = pack_bf2(f0.z, f1.z);
      *(unsigned int*)&Bl[buf][(bc0 + 3) * 40 + 2 * kp] = pack_bf2(f0.w, f1.w);
    }
  };

  float4 Br[4], Br2[4];
  loadB(0, Br);
  stageA(0, 0);

  for (int kc = 0; kc < NC; kc++) {
    const int buf = kc & 1;
    storeB(buf, Br);
    if (kc + 1 < NC) loadB(kc + 1, Br2);
    wait_async();
    __syncthreads();
    if (kc + 1 < NC) stageA(kc + 1, buf ^ 1);

    const unsigned short* Ac = Al[buf];
    const unsigned short* Bc = Bl[buf];
    v16bf fa[2], fb[4];
#pragma unroll
    for (int a = 0; a < 2; a++) {
      // A frag: lanes<16 K{0-7,16-23}, lanes>=16 K{8-15,24-31}
      const unsigned short* base = &Ac[(m_w + a * 16 + r) * 40 + hl * 8];
      fa[a] = ld_frag(base, base + 16);
    }
#pragma unroll
    for (int bi = 0; bi < 4; bi++) {
      // B frag: lane n holds column n; lanes<16 K{0-7,16-23}, lanes>=16 K{8-15,24-31}
      const unsigned short* base = &Bc[(n_w + bi * 16 + r) * 40 + hl * 8];
      fb[bi] = ld_frag(base, base + 16);
    }
#pragma unroll
    for (int a = 0; a < 2; a++)
#pragma unroll
      for (int bi = 0; bi < 4; bi++)
        acc[a][bi] = wmma_bf16(fa[a], fb[bi], acc[a][bi]);

#pragma unroll
    for (int i = 0; i < 4; i++) Br[i] = Br2[i];
  }

  // Epilogue. C layout: row = v + 8*hl, col = lane&15 (within 16x16 tile).
#pragma unroll
  for (int a = 0; a < 2; a++) {
#pragma unroll
    for (int bi = 0; bi < 4; bi++) {
      const int col = n_base + n_w + bi * 16 + r;
      const float bv = bias[col];
#pragma unroll
      for (int v = 0; v < 8; v++) {
        const int row = m_base + m_w + a * 16 + v + 8 * hl;
        const float val = acc[a][bi][v] + bv;
        const size_t idx = (size_t)row * N + col;
        if (EPI == 0) {
          ((unsigned short*)outp)[idx] = f2bf(val);
        } else if (EPI == 1) {
          const float g = 0.5f * val * (1.0f + erff(val * 0.70710678118654752f));
          ((unsigned short*)outp)[idx] = f2bf(g);
        } else {
          float* o = (float*)outp;
          o[idx] = o[idx] + val;  // residual add
        }
      }
    }
  }
}

// ---------------------------------------------------------------------------
// Flash attention. Grid (S/64, H, B); 128 threads = 4 waves, each wave owns
// 16 q-rows. qkv is bf16 [4096, 2304]; per head h: q @ h*192, k @ +64, v @ +128.
// Online softmax; Q@K^T and P@V on WMMA; K/V double-buffered, K async-staged.
// ---------------------------------------------------------------------------
__global__ __launch_bounds__(128) void attn_kernel(
    const unsigned short* __restrict__ qkv, unsigned short* __restrict__ attnb) {
  __shared__ unsigned short Kl[2][64 * 72];   // K block, row-major (key, hd)
  __shared__ unsigned short Vt[2][64 * 72];   // V block, transposed (hd, key)
  __shared__ unsigned short Pl[4 * 16 * 72];  // per-wave P staging

  const int tid = threadIdx.x;
  const int lane = tid & 31, w = tid >> 5;
  const int r = lane & 15, hl = lane >> 4;
  const int qblk = blockIdx.x, h = blockIdx.y, bb = blockIdx.z;
  const size_t rowbase = (size_t)(bb * S_ + qblk * 64);
  const size_t batchbase = (size_t)(bb * S_);
  const int NB = S_ / 64;

  // Q fragments for this wave's 16 rows (kept in registers for whole kernel)
  v16bf fq[2];
  {
    const unsigned short* qb =
        qkv + (rowbase + w * 16 + r) * (3 * D_) + h * (3 * HD_);
#pragma unroll
    for (int c = 0; c < 2; c++)
      fq[c] = ld_frag(qb + c * 32 + hl * 8, qb + c * 32 + hl * 8 + 16);
  }

  auto stageK = [&](int kb, int buf) {
    const int row = tid >> 1, seg = tid & 1;
    const unsigned short* g = qkv + (batchbase + kb * 64 + row) * (3 * D_) +
                              h * (3 * HD_) + HD_ + seg * 32;
    unsigned short* l = &Kl[buf][row * 72 + seg * 32];
    async_b128(g, l);
    async_b128(g + 8, l + 8);
    async_b128(g + 16, l + 16);
    async_b128(g + 24, l + 24);
  };
  auto loadV = [&](int kb, uint4* d) {
#pragma unroll
    for (int it = 0; it < 2; it++) {
      const int task = tid + it * 128;
      const int kp = task & 31, cg = task >> 5;
      const unsigned short* g0 = qkv + (batchbase + kb * 64 + 2 * kp) * (3 * D_) +
                                 h * (3 * HD_) + 2 * HD_ + cg * 8;
      d[2 * it]     = *(const uint4*)g0;
      d[2 * it + 1] = *(const uint4*)(g0 + 3 * D_);
    }
  };
  auto storeV = [&](int buf, const uint4* d) {
#pragma unroll
    for (int it = 0; it < 2; it++) {
      const int task = tid + it * 128;
      const int kp = task & 31, cg = task >> 5;
      const unsigned short* s0 = (const unsigned short*)&d[2 * it];
      const unsigned short* s1 = (const unsigned short*)&d[2 * it + 1];
#pragma unroll
      for (int j = 0; j < 8; j++) {
        unsigned int u = (unsigned int)s0[j] | ((unsigned int)s1[j] << 16);
        *(unsigned int*)&Vt[buf][(cg * 8 + j) * 72 + 2 * kp] = u;
      }
    }
  };

  float mrow[8], lrow[8];
  v8f O[4];
#pragma unroll
  for (int v = 0; v < 8; v++) { mrow[v] = -1e30f; lrow[v] = 0.0f; }
#pragma unroll
  for (int s = 0; s < 4; s++)
#pragma unroll
    for (int v = 0; v < 8; v++) O[s][v] = 0.0f;

  unsigned short* Pm = &Pl[w * 16 * 72];

  uint4 Vr[4], Vr2[4];
  loadV(0, Vr);
  stageK(0, 0);

  for (int kb = 0; kb < NB; kb++) {
    const int buf = kb & 1;
    storeV(buf, Vr);
    if (kb + 1 < NB) loadV(kb + 1, Vr2);
    wait_async();
    __syncthreads();
    if (kb + 1 < NB) stageK(kb + 1, buf ^ 1);

    const unsigned short* Kc = Kl[buf];
    const unsigned short* Vc = Vt[buf];

    // S = Q @ K^T  (row-major K == column-major B)
    v8f sacc[4];
#pragma unroll
    for (int s = 0; s < 4; s++)
#pragma unroll
      for (int v = 0; v < 8; v++) sacc[s][v] = 0.0f;
#pragma unroll
    for (int c = 0; c < 2; c++) {
#pragma unroll
      for (int s = 0; s < 4; s++) {
        const unsigned short* base = &Kc[(s * 16 + r) * 72 + c * 32 + hl * 8];
        v16bf fk = ld_frag(base, base + 16);
        sacc[s] = wmma_bf16(fq[c], fk, sacc[s]);
      }
    }

    // Online softmax update per row (row = v + 8*hl; reduce over 16-lane half)
#pragma unroll
    for (int v = 0; v < 8; v++) {
      float x0 = sacc[0][v] * 0.125f, x1 = sacc[1][v] * 0.125f;
      float x2 = sacc[2][v] * 0.125f, x3 = sacc[3][v] * 0.125f;
      float mx = fmaxf(fmaxf(x0, x1), fmaxf(x2, x3));
      mx = fmaxf(mx, __shfl_xor(mx, 1));
      mx = fmaxf(mx, __shfl_xor(mx, 2));
      mx = fmaxf(mx, __shfl_xor(mx, 4));
      mx = fmaxf(mx, __shfl_xor(mx, 8));
      const float mn = fmaxf(mrow[v], mx);
      const float corr = __expf(mrow[v] - mn);
      const float p0 = __expf(x0 - mn), p1 = __expf(x1 - mn);
      const float p2 = __expf(x2 - mn), p3 = __expf(x3 - mn);
      float ls = p0 + p1 + p2 + p3;
      ls += __shfl_xor(ls, 1);
      ls += __shfl_xor(ls, 2);
      ls += __shfl_xor(ls, 4);
      ls += __shfl_xor(ls, 8);
      lrow[v] = lrow[v] * corr + ls;
      mrow[v] = mn;
#pragma unroll
      for (int s = 0; s < 4; s++) O[s][v] *= corr;
      const int prow = v + 8 * hl;
      Pm[prow * 72 + r]      = f2bf(p0);
      Pm[prow * 72 + 16 + r] = f2bf(p1);
      Pm[prow * 72 + 32 + r] = f2bf(p2);
      Pm[prow * 72 + 48 + r] = f2bf(p3);
    }

    // O += P @ V  (Vt column-major over keys)
#pragma unroll
    for (int c = 0; c < 2; c++) {
      const unsigned short* pb = &Pm[r * 72 + c * 32 + hl * 8];
      v16bf fp = ld_frag(pb, pb + 16);
#pragma unroll
      for (int s = 0; s < 4; s++) {
        const unsigned short* vb = &Vc[(s * 16 + r) * 72 + c * 32 + hl * 8];
        v16bf fv = ld_frag(vb, vb + 16);
        O[s] = wmma_bf16(fp, fv, O[s]);
      }
    }

#pragma unroll
    for (int i = 0; i < 4; i++) Vr[i] = Vr2[i];
  }

  // Normalize and write bf16: attnb[token][h*64 + hd]
#pragma unroll
  for (int v = 0; v < 8; v++) {
    const float inv = 1.0f / lrow[v];
    const size_t orow = rowbase + w * 16 + v + 8 * hl;
#pragma unroll
    for (int s = 0; s < 4; s++)
      attnb[orow * D_ + h * HD_ + s * 16 + r] = f2bf(O[s][v] * inv);
  }
}

// ---------------------------------------------------------------------------
extern "C" void kernel_launch(void* const* d_in, const int* in_sizes, int n_in,
                              void* d_out, int out_size, void* d_ws,
                              size_t ws_size, hipStream_t stream) {
  (void)in_sizes; (void)n_in; (void)out_size; (void)ws_size;
  const float* x     = (const float*)d_in[0];
  const float* qkv_w = (const float*)d_in[1];
  const float* qkv_b = (const float*)d_in[2];
  const float* out_w = (const float*)d_in[3];
  const float* out_b = (const float*)d_in[4];
  const float* ln1_w = (const float*)d_in[5];
  const float* ln1_b = (const float*)d_in[6];
  const float* fc1_w = (const float*)d_in[7];
  const float* fc1_b = (const float*)d_in[8];
  const float* fc2_w = (const float*)d_in[9];
  const float* fc2_b = (const float*)d_in[10];
  const float* ln2_w = (const float*)d_in[11];
  const float* ln2_b = (const float*)d_in[12];

  float* xres = (float*)d_out;  // fp32 residual stream lives in d_out

  unsigned short* hbf   = (unsigned short*)d_ws;              // 4096 x 768
  unsigned short* qkvb  = hbf + (size_t)MTOK * D_;            // 4096 x 2304
  unsigned short* attnb = qkvb + (size_t)MTOK * 3 * D_;       // 4096 x 768
  unsigned short* h1b   = attnb + (size_t)MTOK * D_;          // 4096 x 3072

  (void)hipMemcpyAsync(xres, x, (size_t)MTOK * D_ * sizeof(float),
                       hipMemcpyDeviceToDevice, stream);

  for (int l = 0; l < L_; l++) {
    const float* qw  = qkv_w + (size_t)l * D_ * 3 * D_;
    const float* qb  = qkv_b + (size_t)l * 3 * D_;
    const float* ow  = out_w + (size_t)l * D_ * D_;
    const float* ob  = out_b + (size_t)l * D_;
    const float* l1w = ln1_w + (size_t)l * D_;
    const float* l1b = ln1_b + (size_t)l * D_;
    const float* f1w = fc1_w + (size_t)l * D_ * I_;
    const float* f1b = fc1_b + (size_t)l * I_;
    const float* f2w = fc2_w + (size_t)l * I_ * D_;
    const float* f2b = fc2_b + (size_t)l * D_;
    const float* l2w = ln2_w + (size_t)l * D_;
    const float* l2b = ln2_b + (size_t)l * D_;

    // h = LN1(x) -> bf16
    ln_kernel<<<MTOK, 256, 0, stream>>>(xres, l1w, l1b, hbf);
    // qkv = h @ qkv_w + qkv_b  -> bf16 [4096, 2304]
    gemm_kernel<0><<<dim3(3 * D_ / 128, MTOK / 128), 256, 0, stream>>>(
        hbf, qw, qb, qkvb, 3 * D_, D_);
    // attention -> bf16 [4096, 768]
    attn_kernel<<<dim3(S_ / 64, H_, B_), 128, 0, stream>>>(qkvb, attnb);
    // x += attn @ out_w + out_b
    gemm_kernel<2><<<dim3(D_ / 128, MTOK / 128), 256, 0, stream>>>(
        attnb, ow, ob, xres, D_, D_);
    // h = LN2(x) -> bf16
    ln_kernel<<<MTOK, 256, 0, stream>>>(xres, l2w, l2b, hbf);
    // h1 = gelu(h @ fc1_w + fc1_b) -> bf16 [4096, 3072]
    gemm_kernel<1><<<dim3(I_ / 128, MTOK / 128), 256, 0, stream>>>(
        hbf, f1w, f1b, h1b, I_, D_);
    // x += h1 @ fc2_w + fc2_b
    gemm_kernel<2><<<dim3(D_ / 128, MTOK / 128), 256, 0, stream>>>(
        h1b, f2w, f2b, xres, D_, I_);
  }
}
